// SelectiveSSM_13434657702438
// MI455X (gfx1250) — matrix operand order repaired and checked by reading, not compiled, not run on hardware
//
#include <hip/hip_runtime.h>
#include <hip/hip_bf16.h>
#include <stdint.h>

typedef __attribute__((ext_vector_type(16))) __bf16 v16bf;
typedef __attribute__((ext_vector_type(8)))  float  v8f;

// Problem dims (fixed by reference)
#define BDIM 8
#define LDIM 4096
#define DDIM 1024
#define MTOT (BDIM * LDIM)   // 32768 rows

__device__ __forceinline__ unsigned short f2bf(float f) {
    union { float f; unsigned u; } c; c.f = f;
    unsigned u = c.u;
    // round-to-nearest-even truncation to bf16
    unsigned r = (u + 0x7FFFu + ((u >> 16) & 1u)) >> 16;
    return (unsigned short)r;
}

// ---------------------------------------------------------------------------
// CDNA5 async global->LDS DMA (ASYNCcnt-tracked, no VGPR staging).
// vDST operand = wave-relative LDS byte address (= low 32 bits of the generic
// pointer to a __shared__ object; LDS aperture lives in addr[63:32]).
// ---------------------------------------------------------------------------
__device__ __forceinline__ void async_ld_b128(unsigned lds_addr, const void* gaddr) {
    asm volatile("global_load_async_to_lds_b128 %0, %1, off"
                 :: "v"(lds_addr), "v"(gaddr) : "memory");
}
template <int N>
__device__ __forceinline__ void wait_async_le() {
    asm volatile("s_wait_asynccnt %0" :: "n"(N) : "memory");
}
__device__ __forceinline__ unsigned lds_off(const void* p) {
    return (unsigned)(uintptr_t)p;
}
// Opaque pass-through: stops LICM from rewriting running pointers as
// hoisted base+const (which exploded VGPR live ranges).
__device__ __forceinline__ void opaque(const unsigned short*& p) {
    asm("" : "+v"(p));
}

// ---------------------------------------------------------------------------
// Kernel 0: Wout fp32 -> bf16 (tiny, 2MB output, L2 resident afterwards)
// ---------------------------------------------------------------------------
__global__ void wconv_kernel(const float* __restrict__ W,
                             unsigned short* __restrict__ Wb) {
    int i = (blockIdx.x * 256 + threadIdx.x) * 4;
    float4 f = *(const float4*)(W + i);
    union { unsigned short s[4]; uint2 u; } p;
    p.s[0] = f2bf(f.x); p.s[1] = f2bf(f.y);
    p.s[2] = f2bf(f.z); p.s[3] = f2bf(f.w);
    *(uint2*)(Wb + i) = p.u;
}

// ---------------------------------------------------------------------------
// Kernel 1: gate  g[row] = sigmoid(x[row,:] . Wg + bg)
// One wave32 per row; Wg staged in LDS; float4 loads; xor-shuffle reduction.
// ---------------------------------------------------------------------------
__global__ __launch_bounds__(256) void gate_kernel(const float* __restrict__ x,
                            const float* __restrict__ Wg,
                            const float* __restrict__ bg,
                            float* __restrict__ g) {
    __shared__ float4 sWg[DDIM / 4];
    const int tid = threadIdx.x;
    const float4* wg4 = (const float4*)Wg;
    for (int i = tid; i < DDIM / 4; i += 256) sWg[i] = wg4[i];
    __syncthreads();

    const int wave = tid >> 5, lane = tid & 31;
    const int row  = blockIdx.x * 8 + wave;       // 4096 blocks * 8 waves = 32768 rows
    const float4* xr = (const float4*)(x + (size_t)row * DDIM);

    float acc = 0.f;
#pragma unroll
    for (int j = 0; j < 8; ++j) {
        float4 a = xr[lane + 32 * j];
        float4 w = sWg[lane + 32 * j];
        acc += a.x * w.x + a.y * w.y + a.z * w.z + a.w * w.w;
    }
#pragma unroll
    for (int off = 16; off > 0; off >>= 1) acc += __shfl_xor(acc, off, 32);
    if (lane == 0) g[row] = 1.f / (1.f + __expf(-(acc + bg[0])));
}

// ---------------------------------------------------------------------------
// Kernel 2: sequential scan per channel, emit H as bf16 row-major [M][D]
// h_t = h_{t-1} + g_t * (x_t - h_{t-1})
// ---------------------------------------------------------------------------
__global__ __launch_bounds__(256) void scan_kernel(const float* __restrict__ x,
                            const float* __restrict__ g,
                            unsigned short* __restrict__ Hbf) {
    const int b = blockIdx.x >> 2;                       // 8 batches
    const int d = ((blockIdx.x & 3) << 8) + threadIdx.x; // 4*256 channels per batch
    const float* xb = x + ((size_t)b * LDIM) * DDIM + d;
    const float* gb = g + (size_t)b * LDIM;              // uniform per block -> scalar loads
    unsigned short* hb = Hbf + ((size_t)b * LDIM) * DDIM + d;

    float h = 0.f;
    for (int t = 0; t < LDIM; t += 4) {
        float xv[4], gv[4];
#pragma unroll
        for (int u = 0; u < 4; ++u) {
            xv[u] = xb[(size_t)(t + u) * DDIM];
            gv[u] = gb[t + u];
        }
        if (t + 16 < LDIM) __builtin_prefetch(&xb[(size_t)(t + 16) * DDIM], 0, 0);
#pragma unroll
        for (int u = 0; u < 4; ++u) {
            h = fmaf(gv[u], xv[u] - h, h);
            hb[(size_t)(t + u) * DDIM] = f2bf(h);
        }
    }
}

// ---------------------------------------------------------------------------
// Kernel 3: out[M,N] = Hbf[M,K] * Woutbf[N,K]^T + bout[N]   (M=32768,N=K=1024)
// WMMA v_wmma_f32_16x16x32_bf16. Block tile 128x64, 8 waves, each wave 2x2
// 16x16 subtiles, K-step 32. Triple-buffered LDS fed by
// GLOBAL_LOAD_ASYNC_TO_LDS_B128 with 2 tiles in flight (ASYNCcnt<=3 waits).
// Fully unrolled K loop; running DMA pointers kept opaque so they stay as
// 3 register pairs + one +64B bump per stream per step.
//
// Fragment layouts per CDNA5 ISA 7.12.2 (wave32):
//  A (16x32 bf16): lane = half*16 + m ; element j -> K = (j&7) + 8*half + 16*(j>>3)
//  B (32x16 bf16): lane = half*16 + n ; element j -> K = (j&7) + 8*half + 16*(j>>3) (same as A)
//  C/D (16x16 f32): vgpr r, lane -> M = r + 8*half, N = lane&15
// B staged as Wout rows (n-major) so each lane's K-column is contiguous.
// ---------------------------------------------------------------------------
#define TK 32
#define LDA_PAD 40   // 32 + 8 halfwords (80B row stride) to spread LDS banks

__global__ __launch_bounds__(256) void gemm_kernel(const unsigned short* __restrict__ A,   // Hbf [M][1024]
                            const unsigned short* __restrict__ Bw,  // Woutbf [1024][1024]
                            const float* __restrict__ bout,
                            float* __restrict__ out) {
    __shared__ __align__(16) unsigned short ldsA[3][128 * LDA_PAD]; // 30KB
    __shared__ __align__(16) unsigned short ldsB[3][64 * LDA_PAD];  // 15KB

    const int tid  = threadIdx.x;
    const int lane = tid & 31, wv = tid >> 5;
    const int half = lane >> 4, ml = lane & 15;
    const size_t rowA0 = (size_t)blockIdx.x * 128;  // 256 M-tiles
    const int    colB0 = blockIdx.y * 64;           // 16 N-tiles

    // ---- per-thread DMA mapping: 16B chunks; A = 128 rows x 4 chunks (2/thread),
    //      B = 64 rows x 4 chunks (1/thread). Running pointers, +TK per tile. ----
    const int arow = tid >> 2, aseg = tid & 3;
    const unsigned short* pA0 = A  + (rowA0 + arow) * DDIM + aseg * 8;
    const unsigned short* pA1 = A  + (rowA0 + arow + 64) * DDIM + aseg * 8;
    const unsigned short* pB  = Bw + (size_t)(colB0 + arow) * DDIM + aseg * 8;

    // fixed LDS destinations (per thread, per buffer)
    const unsigned la0 = lds_off(&ldsA[0][arow * LDA_PAD + aseg * 8]);
    const unsigned la1 = lds_off(&ldsA[0][(arow + 64) * LDA_PAD + aseg * 8]);
    const unsigned lb0 = lds_off(&ldsB[0][arow * LDA_PAD + aseg * 8]);
    const unsigned strA = sizeof(ldsA[0]);
    const unsigned strB = sizeof(ldsB[0]);

    const int NK = DDIM / TK;  // 32

    auto issue_tile = [&](int nb) {
        async_ld_b128(la0 + nb * strA, pA0);
        async_ld_b128(la1 + nb * strA, pA1);
        async_ld_b128(lb0 + nb * strB, pB);
        pA0 += TK; pA1 += TK; pB += TK;
        opaque(pA0); opaque(pA1); opaque(pB);
    };

    v8f acc[2][2] = {};
    const int mbase = (wv & 3) * 32;   // 4 waves along M
    const int nbase = (wv >> 2) * 32;  // 2 waves along N

    union FragU { uint4 q[2]; v16bf v; };

    // prologue: two tiles in flight
    issue_tile(0);
    issue_tile(1);

#pragma unroll
    for (int kt = 0; kt < NK; ++kt) {
        // wait for tile kt (oldest): at most the 3 ops of tile kt+1 may remain.
        if (kt + 1 < NK) wait_async_le<3>(); else wait_async_le<0>();
        __syncthreads();
        // kick DMA for tile kt+2 into buffer (kt+2)%3: that buffer's reads
        // finished at iteration kt-1, sequenced before this barrier.
        if (kt + 2 < NK) issue_tile((kt + 2) % 3);

        const int buf = kt % 3;
        FragU afr[2], bfr[2];
#pragma unroll
        for (int i = 0; i < 2; ++i) {
            const int r = mbase + 16 * i + ml;
            afr[i].q[0] = *(const uint4*)&ldsA[buf][r * LDA_PAD + 8 * half];
            afr[i].q[1] = *(const uint4*)&ldsA[buf][r * LDA_PAD + 8 * half + 16];
            const int n = nbase + 16 * i + ml;
            bfr[i].q[0] = *(const uint4*)&ldsB[buf][n * LDA_PAD + 8 * half];
            bfr[i].q[1] = *(const uint4*)&ldsB[buf][n * LDA_PAD + 8 * half + 16];
        }

#pragma unroll
        for (int i = 0; i < 2; ++i)
#pragma unroll
            for (int j = 0; j < 2; ++j)
                acc[i][j] = __builtin_amdgcn_wmma_f32_16x16x32_bf16(
                    false, afr[i].v, false, bfr[j].v,
                    (short)0, acc[i][j], false, false);
    }

    // epilogue: D layout M = r + 8*half, N = ml
#pragma unroll
    for (int i = 0; i < 2; ++i) {
#pragma unroll
        for (int j = 0; j < 2; ++j) {
            const size_t gm = rowA0 + mbase + 16 * i + 8 * half;
            const int    gn = colB0 + nbase + 16 * j + ml;
            const float  bb = bout[gn];
#pragma unroll
            for (int r = 0; r < 8; ++r)
                out[(gm + r) * DDIM + gn] = acc[i][j][r] + bb;
        }
    }
}

// ---------------------------------------------------------------------------
// Host launcher
// ---------------------------------------------------------------------------
extern "C" void kernel_launch(void* const* d_in, const int* in_sizes, int n_in,
                              void* d_out, int out_size, void* d_ws, size_t ws_size,
                              hipStream_t stream) {
    const float* x    = (const float*)d_in[0];   // [8,4096,1024]
    const float* Wg   = (const float*)d_in[1];   // [1,1024]
    const float* bg   = (const float*)d_in[2];   // [1]
    const float* Wout = (const float*)d_in[3];   // [1024,1024]
    const float* bout = (const float*)d_in[4];   // [1024]
    float* out = (float*)d_out;

    // workspace layout
    char* ws = (char*)d_ws;
    float*          g      = (float*)ws;                               // 32768 * 4   = 128KB
    unsigned short* Woutbf = (unsigned short*)(ws + 32768 * 4);        // 1M * 2      = 2MB
    unsigned short* Hbf    = (unsigned short*)(ws + 32768 * 4 + (size_t)DDIM * DDIM * 2); // 64MB

    wconv_kernel<<<(DDIM * DDIM) / (256 * 4), 256, 0, stream>>>(Wout, Woutbf);
    gate_kernel<<<MTOT / 8, 256, 0, stream>>>(x, Wg, bg, g);
    scan_kernel<<<BDIM * 4, 256, 0, stream>>>(x, g, Hbf);
    dim3 ggrid(MTOT / 128, DDIM / 64, 1);
    gemm_kernel<<<ggrid, 256, 0, stream>>>(Hbf, Woutbf, bout, out);
}
